// MusicVAE_38929583570982
// MI455X (gfx1250) — matrix operand order repaired and checked by reading, not compiled, not run on hardware
//
#include <hip/hip_runtime.h>
#include <hip/hip_bf16.h>

typedef __bf16 bf16;
typedef __attribute__((ext_vector_type(16))) __bf16 v16bf;
typedef __attribute__((ext_vector_type(8)))  float v8f;
typedef __attribute__((ext_vector_type(4)))  unsigned int uint4v;

// ---------------- model dims ----------------
#define B_    512
#define T_    272
#define NC_   96
#define H_    512
#define G_    2048      // 4*H
#define LZ_   256
#define NB_   34
#define CHUNK_ 68       // T/4
#define MD_   ((long)NB_ * B_)   // 17408

// =====================================================================
// WMMA GEMM:  C[M,N] (+)= A[M,K] * W[N,K]^T  (+ bias[n]) (+ Cin)
// A rows addressed two-level: addr = A + (m>>9)*a_hi + (m&511)*a_lo
// (handles time-reversed sequences via negative a_hi). Same for C/Cin.
// A, W bf16; accumulate fp32; output fp32 and/or bf16.
// Block: 256 thr = 8 waves; tile M=64 x N=256; wave tile 32x64 (2x4 WMMA).
// Double-buffered LDS: tile k+1 is fetched to VGPRs while tile k's WMMAs
// run, then committed to the alternate buffer; one barrier per K-tile.
// =====================================================================
#define LDSS 40  // bf16 elems per LDS row (80B: 16B-aligned, low bank conflicts)

__global__ __launch_bounds__(256) void k_gemm(
    const bf16* __restrict__ A, long a_lo, long a_hi,
    const bf16* __restrict__ W, int K, int N,
    float* __restrict__ Cf, bf16* __restrict__ Cb, long c_lo, long c_hi,
    const float* __restrict__ CinF, const bf16* __restrict__ CinB,
    const float* __restrict__ bias)
{
    __shared__ bf16 As[2][64 * LDSS];
    __shared__ bf16 Bs[2][256 * LDSS];
    const int tid  = threadIdx.x;
    const int lane = tid & 31;
    const int wv   = tid >> 5;
    const int wm   = wv & 1;   // 0..1  (M within block)
    const int wn   = wv >> 1;  // 0..3  (N within block)
    const long m0  = (long)blockIdx.y * 64;
    const long n0  = (long)blockIdx.x * 256;

    v8f acc[2][4];
#pragma unroll
    for (int i = 0; i < 2; ++i)
#pragma unroll
        for (int j = 0; j < 4; ++j)
#pragma unroll
            for (int r = 0; r < 8; ++r) acc[i][j][r] = 0.0f;

    const int arow = tid >> 2;        // 0..63
    const int acol = (tid & 3) << 3;  // 0,8,16,24 (bf16 elems)
    const int hi   = lane >> 4;
    const int lr   = lane & 15;

    // uniform-per-block guard: only the last N-tile of narrow GEMMs is ragged
    const bool nfull = (n0 + 256 <= (long)N);

    // staging addresses (K-offset added per tile)
    const long am = m0 + arow;
    const bf16* __restrict__ aptr = A + (am >> 9) * a_hi + (am & 511) * a_lo + acol;
    const bf16* __restrict__ wptr = W + (n0 + arow) * (long)K + acol;

    uint4v ra, rb0, rb1, rb2, rb3;

    auto load_tile = [&](int k0) {
        ra = *reinterpret_cast<const uint4v*>(aptr + k0);
        if (nfull) {
            rb0 = *reinterpret_cast<const uint4v*>(wptr + (long)(64 * 0) * K + k0);
            rb1 = *reinterpret_cast<const uint4v*>(wptr + (long)(64 * 1) * K + k0);
            rb2 = *reinterpret_cast<const uint4v*>(wptr + (long)(64 * 2) * K + k0);
            rb3 = *reinterpret_cast<const uint4v*>(wptr + (long)(64 * 3) * K + k0);
        } else {
            rb0[0]=rb0[1]=rb0[2]=rb0[3]=0u; rb1 = rb0; rb2 = rb0; rb3 = rb0;
            if (n0 + arow + 64*0 < (long)N) rb0 = *reinterpret_cast<const uint4v*>(wptr + (long)(64*0)*K + k0);
            if (n0 + arow + 64*1 < (long)N) rb1 = *reinterpret_cast<const uint4v*>(wptr + (long)(64*1)*K + k0);
            if (n0 + arow + 64*2 < (long)N) rb2 = *reinterpret_cast<const uint4v*>(wptr + (long)(64*2)*K + k0);
            if (n0 + arow + 64*3 < (long)N) rb3 = *reinterpret_cast<const uint4v*>(wptr + (long)(64*3)*K + k0);
        }
    };
    auto store_tile = [&](int buf) {
        *reinterpret_cast<uint4v*>(&As[buf][arow * LDSS + acol]) = ra;
        *reinterpret_cast<uint4v*>(&Bs[buf][(arow + 64*0) * LDSS + acol]) = rb0;
        *reinterpret_cast<uint4v*>(&Bs[buf][(arow + 64*1) * LDSS + acol]) = rb1;
        *reinterpret_cast<uint4v*>(&Bs[buf][(arow + 64*2) * LDSS + acol]) = rb2;
        *reinterpret_cast<uint4v*>(&Bs[buf][(arow + 64*3) * LDSS + acol]) = rb3;
    };

    // prologue: stage tile 0
    load_tile(0);
    store_tile(0);
    __syncthreads();

    const int nk = K >> 5;   // K is always a multiple of 32 in this model
    for (int kt = 0; kt < nk; ++kt) {
        const int cur  = kt & 1;
        const bool more = (kt + 1) < nk;
        // kick off next tile's global loads; they retire under the WMMAs
        if (more) load_tile((kt + 1) << 5);

        // ---- fragments (ISA 16-bit A 16x32 / B 32x16 lane layouts) ----
        union FR { uint4v u[2]; v16bf v; };
        FR a0, a1, b[4];
        {
            int r0 = (wm * 32 + lr) * LDSS;
            a0.u[0] = *reinterpret_cast<const uint4v*>(&As[cur][r0 + 8 * hi]);
            a0.u[1] = *reinterpret_cast<const uint4v*>(&As[cur][r0 + 16 + 8 * hi]);
            int r1 = (wm * 32 + 16 + lr) * LDSS;
            a1.u[0] = *reinterpret_cast<const uint4v*>(&As[cur][r1 + 8 * hi]);
            a1.u[1] = *reinterpret_cast<const uint4v*>(&As[cur][r1 + 16 + 8 * hi]);
        }
#pragma unroll
        for (int j = 0; j < 4; ++j) {
            int rb = (wn * 64 + j * 16 + lr) * LDSS + 8 * hi;
            b[j].u[0] = *reinterpret_cast<const uint4v*>(&Bs[cur][rb]);
            b[j].u[1] = *reinterpret_cast<const uint4v*>(&Bs[cur][rb + 16]);
        }
#pragma unroll
        for (int j = 0; j < 4; ++j) {
            acc[0][j] = __builtin_amdgcn_wmma_f32_16x16x32_bf16(
                false, a0.v, false, b[j].v, (short)0, acc[0][j], false, false);
            acc[1][j] = __builtin_amdgcn_wmma_f32_16x16x32_bf16(
                false, a1.v, false, b[j].v, (short)0, acc[1][j], false, false);
        }

        // commit next tile to the alternate buffer (safe: last reader of that
        // buffer finished before the previous barrier)
        if (more) store_tile(cur ^ 1);
        __syncthreads();
    }

    // ---- epilogue: bias + Cin accumulate + fp32/bf16 store ----
#pragma unroll
    for (int mi = 0; mi < 2; ++mi)
#pragma unroll
        for (int j = 0; j < 4; ++j) {
            long n = n0 + wn * 64 + j * 16 + lr;
            if (n >= (long)N) continue;
            float bv = bias ? bias[n] : 0.0f;
#pragma unroll
            for (int r = 0; r < 8; ++r) {
                long m  = m0 + wm * 32 + mi * 16 + 8 * hi + r;
                long ro = (m >> 9) * c_hi + (m & 511) * c_lo + n;
                float v = acc[mi][j][r] + bv;
                if (CinF)       v += CinF[ro];
                else if (CinB)  v += (float)CinB[ro];
                if (Cf) Cf[ro] = v;
                if (Cb) Cb[ro] = (bf16)v;
            }
        }
}

// =====================================================================
// elementwise kernels
// =====================================================================
__global__ void k_cvt(bf16* d, const float* s, long n) {
    long i = (long)blockIdx.x * 256 + threadIdx.x;
    if (i < n) d[i] = (bf16)s[i];
}
__global__ void k_add(float* d, const float* a, const float* b, long n) {
    long i = (long)blockIdx.x * 256 + threadIdx.x;
    if (i < n) d[i] = a[i] + b[i];
}
__global__ void k_zerof(float* p, long n) {
    long i = (long)blockIdx.x * 256 + threadIdx.x;
    if (i < n) p[i] = 0.0f;
}
__global__ void k_zerob(bf16* p, long n) {
    long i = (long)blockIdx.x * 256 + threadIdx.x;
    if (i < n) p[i] = (bf16)0.0f;
}
__global__ void k_copyf(float* d, const float* s, long n) {
    long i = (long)blockIdx.x * 256 + threadIdx.x;
    if (i < n) d[i] = s[i];
}
__global__ void k_copyb(bf16* d, const bf16* s, long n) {
    long i = (long)blockIdx.x * 256 + threadIdx.x;
    if (i < n) d[i] = s[i];
}
// copy [M,512] contiguous bf16 into strided columns of hidden buffer
__global__ void k_copy_rows_b(const bf16* s, bf16* d, long ldst, long n) {
    long i = (long)blockIdx.x * 256 + threadIdx.x;
    if (i < n) d[(i >> 9) * ldst + (i & 511)] = s[i];
}
// x [B,T,NC] fp32 -> xt [T,B,NC] bf16
__global__ void k_xt(const float* x, bf16* xt, long n) {
    long i = (long)blockIdx.x * 256 + threadIdx.x;
    if (i >= n) return;
    long k = i % NC_;
    long t = (i / NC_) % T_;
    long b = i / ((long)NC_ * T_);
    xt[(t * B_ + b) * NC_ + k] = (bf16)x[i];
}
__device__ __forceinline__ float sigm(float x) { return 1.0f / (1.0f + expf(-x)); }

// LSTM gates: G[m,2048] = [i|f|g|o]; updates c,h; emits bf16 h copies
__global__ void k_gate(const float* __restrict__ Gm, const float* __restrict__ ci,
                       float* co, float* ho, bf16* hb1, bf16* hb2, long hb2_stride, long n) {
    long idx = (long)blockIdx.x * 256 + threadIdx.x;   // over M*512
    if (idx >= n) return;
    long m = idx >> 9;
    int  j = (int)(idx & 511);
    const float* g = Gm + m * (long)G_;
    float gi = sigm(g[j]);
    float gf = sigm(g[512 + j]);
    float gg = tanhf(g[1024 + j]);
    float go = sigm(g[1536 + j]);
    float c  = gf * ci[idx] + gi * gg;
    float h  = go * tanhf(c);
    co[idx] = c;
    ho[idx] = h;
    hb1[idx] = (bf16)h;
    if (hb2) hb2[m * hb2_stride + j] = (bf16)h;
}
// z = eps * exp(0.5*lv) + mu  (bf16 out for downstream GEMMs)
__global__ void k_reparam(const float* mu, const float* lv, const float* eps, bf16* zb, long n) {
    long i = (long)blockIdx.x * 256 + threadIdx.x;
    if (i < n) zb[i] = (bf16)(eps[i] * expf(0.5f * lv[i]) + mu[i]);
}
// dec_x[m,0:512]=emb_bf[m]; dec_x[m,512:608]=teacher (shifted x), bf16
__global__ void k_pack_decx(const bf16* embB, const float* x, bf16* dx, int step, long n) {
    long i = (long)blockIdx.x * 256 + threadIdx.x;
    if (i >= n) return;
    long m = i / 608;
    int  k = (int)(i % 608);
    bf16 v;
    if (k < 512) {
        v = embB[m * 512 + k];
    } else {
        int b = (int)(m & 511), bar = (int)(m >> 9);
        int gt = bar * 8 + step;
        v = (gt == 0) ? (bf16)0.0f
                      : (bf16)x[((long)b * T_ + (gt - 1)) * NC_ + (k - 512)];
    }
    dx[i] = v;
}
// dec_h0/c0 [NB,2,B,512] -> per-layer [NB*B,512] (+bf16 copies for h)
__global__ void k_unpack_dec(const float* src, float* d0, float* d1, bf16* d0b, bf16* d1b, long n) {
    long i = (long)blockIdx.x * 256 + threadIdx.x;
    if (i >= n) return;
    int  k   = (int)(i & 511);
    int  b   = (int)((i >> 9) & 511);
    int  l   = (int)((i >> 18) & 1);
    long bar = i >> 19;
    float v  = src[i];
    long dst = (bar * B_ + b) * (long)H_ + k;
    if (l == 0) { d0[dst] = v; if (d0b) d0b[dst] = (bf16)v; }
    else        { d1[dst] = v; if (d1b) d1b[dst] = (bf16)v; }
}
// wave-per-row softmax + log_softmax over NC=96 = 3 elems/lane (wave32)
__global__ __launch_bounds__(256) void k_softmax(const float* Lt, float* sm, float* lp, int step) {
    long gid = (long)blockIdx.x * 256 + threadIdx.x;
    long wid = gid >> 5;
    int lane = (int)(gid & 31);
    if (wid >= MD_) return;
    int b = (int)(wid & 511), bar = (int)(wid >> 9);
    const float* row = Lt + wid * NC_;
    float v0 = row[lane], v1 = row[lane + 32], v2 = row[lane + 64];
    float mx = fmaxf(fmaxf(v0, v1), v2);
    for (int o = 16; o; o >>= 1) mx = fmaxf(mx, __shfl_xor(mx, o, 32));
    float e0 = expf(v0 - mx), e1 = expf(v1 - mx), e2 = expf(v2 - mx);
    float s = e0 + e1 + e2;
    for (int o = 16; o; o >>= 1) s += __shfl_xor(s, o, 32);
    float inv = 1.0f / s, ls = logf(s);
    long t = (long)bar * 8 + step;
    float* smr = sm + ((long)b * T_ + t) * NC_;
    float* lpr = lp + ((long)b * T_ + t) * NC_;
    smr[lane]      = e0 * inv;  smr[lane + 32] = e1 * inv;  smr[lane + 64] = e2 * inv;
    lpr[lane]      = v0 - mx - ls;
    lpr[lane + 32] = v1 - mx - ls;
    lpr[lane + 64] = v2 - mx - ls;
}

// =====================================================================
// host driver
// =====================================================================
static inline dim3 g1(long n) { return dim3((unsigned)((n + 255) / 256)); }

extern "C" void kernel_launch(void* const* d_in, const int* in_sizes, int n_in,
                              void* d_out, int out_size, void* d_ws, size_t ws_size,
                              hipStream_t stream) {
    (void)in_sizes; (void)n_in; (void)out_size; (void)ws_size;
    const float* x    = (const float*)d_in[0];
    // d_in[1] = length (unused: full lengths)
    const float* eps  = (const float*)d_in[2];
    const float* dh0i = (const float*)d_in[3];
    const float* dc0i = (const float*)d_in[4];
    auto P = [&](int i) { return (const float*)d_in[i]; };

    // params flattened in setup_inputs() insertion order:
    // enc[0].fwd{Wih,Whh,bih,bhh}, enc[0].bwd, enc[1].fwd, enc[1].bwd,
    // cond[0], cond[1], dec[0], dec[1],
    // h2mu{W,b}, h2lv{W,b}, z2in{W,b}, z2hid{W,b}, out{W,b}
    int q = 5;
    const float* encW[2][2][4];
    for (int l = 0; l < 2; ++l)
        for (int d = 0; d < 2; ++d)
            for (int k = 0; k < 4; ++k) encW[l][d][k] = P(q++);
    const float* condW[2][4];
    for (int l = 0; l < 2; ++l) for (int k = 0; k < 4; ++k) condW[l][k] = P(q++);
    const float* decW[2][4];
    for (int l = 0; l < 2; ++l) for (int k = 0; k < 4; ++k) decW[l][k] = P(q++);
    const float *h2muWf = P(q++), *h2mub = P(q++);
    const float *h2lvWf = P(q++), *h2lvb = P(q++);
    const float *z2inWf = P(q++), *z2inb = P(q++);
    const float *z2hidWf = P(q++), *z2hidb = P(q++);
    const float *outWf = P(q++), *outb = P(q++);

    // ---------- workspace allocator ----------
    char* wsp = (char*)d_ws;
    size_t off = 0;
    auto alloc = [&](size_t bytes) -> void* {
        void* p = wsp + off;
        off = (off + bytes + 255) & ~(size_t)255;
        return p;
    };
    auto cvt = [&](const float* s, long n) -> bf16* {
        bf16* d = (bf16*)alloc((size_t)n * 2);
        k_cvt<<<g1(n), 256, 0, stream>>>(d, s, n);
        return d;
    };
    auto bsum = [&](const float* a, const float* b) -> float* {
        float* d = (float*)alloc((size_t)G_ * 4);
        k_add<<<g1(G_), 256, 0, stream>>>(d, a, b, G_);
        return d;
    };

    // ---------- weight conversion to bf16 (all fit in L2 afterwards) ----------
    int kinL[2] = {NC_, 2 * H_};
    bf16 *eWih[2][2], *eWhh[2][2]; float* eBs[2][2];
    for (int l = 0; l < 2; ++l)
        for (int d = 0; d < 2; ++d) {
            eWih[l][d] = cvt(encW[l][d][0], (long)G_ * kinL[l]);
            eWhh[l][d] = cvt(encW[l][d][1], (long)G_ * H_);
            eBs[l][d]  = bsum(encW[l][d][2], encW[l][d][3]);
        }
    bf16* cWih0 = cvt(condW[0][0], (long)G_ * NC_);
    bf16* cWhh0 = cvt(condW[0][1], (long)G_ * H_);
    float* cBs0 = bsum(condW[0][2], condW[0][3]);
    bf16* cWih1 = cvt(condW[1][0], (long)G_ * H_);
    bf16* cWhh1 = cvt(condW[1][1], (long)G_ * H_);
    float* cBs1 = bsum(condW[1][2], condW[1][3]);
    bf16* dWih0 = cvt(decW[0][0], (long)G_ * (NC_ + H_));
    bf16* dWhh0 = cvt(decW[0][1], (long)G_ * H_);
    float* dBs0 = bsum(decW[0][2], decW[0][3]);
    bf16* dWih1 = cvt(decW[1][0], (long)G_ * H_);
    bf16* dWhh1 = cvt(decW[1][1], (long)G_ * H_);
    float* dBs1 = bsum(decW[1][2], decW[1][3]);
    bf16* Wmu  = cvt(h2muWf, (long)LZ_ * G_);
    bf16* Wlv  = cvt(h2lvWf, (long)LZ_ * G_);
    bf16* Wzi  = cvt(z2inWf, (long)NC_ * LZ_);
    bf16* Wzh  = cvt(z2hidWf, (long)H_ * LZ_);
    bf16* Wout = cvt(outWf, (long)NC_ * H_);

    // ---------- activation buffers ----------
    bf16*  Xt    = (bf16*)alloc((size_t)T_ * B_ * NC_ * 2);          // [T,B,NC]
    bf16*  Y0    = (bf16*)alloc((size_t)T_ * B_ * 1024 * 2);         // enc L0 out (fwd|bwd)
    bf16*  Gx    = (bf16*)alloc((size_t)CHUNK_ * B_ * G_ * 2);       // pre-GEMM chunk
    float* Gstep = (float*)alloc((size_t)B_ * G_ * 4);
    float* hS    = (float*)alloc((size_t)B_ * H_ * 4);
    float* cS    = (float*)alloc((size_t)B_ * H_ * 4);
    bf16*  hSb   = (bf16*)alloc((size_t)B_ * H_ * 2);
    bf16*  hidB  = (bf16*)alloc((size_t)B_ * G_ * 2);                // [B,4*EH]
    bf16*  zbuf  = (bf16*)alloc((size_t)B_ * LZ_ * 2);
    bf16*  zinB  = (bf16*)alloc((size_t)B_ * NC_ * 2);
    float* zhidF = (float*)alloc((size_t)B_ * H_ * 4);
    bf16*  zhidB = (bf16*)alloc((size_t)B_ * H_ * 2);
    float* Gc0   = (float*)alloc((size_t)B_ * G_ * 4);
    float *ch0 = (float*)alloc((size_t)B_ * H_ * 4), *cc0 = (float*)alloc((size_t)B_ * H_ * 4);
    float *ch1 = (float*)alloc((size_t)B_ * H_ * 4), *cc1 = (float*)alloc((size_t)B_ * H_ * 4);
    bf16 *ch0b = (bf16*)alloc((size_t)B_ * H_ * 2), *ch1b = (bf16*)alloc((size_t)B_ * H_ * 2);
    bf16*  embB = (bf16*)alloc((size_t)MD_ * H_ * 2);                // [NB*B,512]
    float *dh0 = (float*)alloc((size_t)MD_ * H_ * 4), *dc0 = (float*)alloc((size_t)MD_ * H_ * 4);
    float *dh1 = (float*)alloc((size_t)MD_ * H_ * 4), *dc1 = (float*)alloc((size_t)MD_ * H_ * 4);
    bf16 *dh0b = (bf16*)alloc((size_t)MD_ * H_ * 2), *dh1b = (bf16*)alloc((size_t)MD_ * H_ * 2);
    bf16*  decX = (bf16*)alloc((size_t)MD_ * (NC_ + H_) * 2);
    float* Gd   = (float*)alloc((size_t)MD_ * G_ * 4);
    float* Lt   = (float*)alloc((size_t)MD_ * NC_ * 4);

    // ---------- output layout (softmax, logp, mu, log_var, x) ----------
    const long SZ = (long)B_ * T_ * NC_;
    float* out  = (float*)d_out;
    float* o_sm = out;
    float* o_lp = out + SZ;
    float* o_mu = out + 2 * SZ;
    float* o_lv = o_mu + (long)B_ * LZ_;
    float* o_x  = o_lv + (long)B_ * LZ_;

    auto gemm = [&](const bf16* A, long alo, long ahi, const bf16* W, int K, int N,
                    float* Cf, bf16* Cb, long clo, long chi,
                    const float* CiF, const bf16* CiB, const float* bias, long M) {
        dim3 grid((unsigned)((N + 255) / 256), (unsigned)(M / 64));
        k_gemm<<<grid, dim3(256), 0, stream>>>(A, alo, ahi, W, K, N, Cf, Cb, clo, chi, CiF, CiB, bias);
    };
    auto gate = [&](const float* Gp, const float* ci, float* co, float* ho,
                    bf16* hb1, bf16* hb2, long st, long M) {
        long n = M * H_;
        k_gate<<<g1(n), 256, 0, stream>>>(Gp, ci, co, ho, hb1, hb2, st, n);
    };

    // ---------- x -> time-major bf16 ----------
    k_xt<<<g1(SZ), 256, 0, stream>>>(x, Xt, SZ);

    // ================= encoder (bidirectional, 2 layers) =================
    for (int l = 0; l < 2; ++l) {
        int kin = kinL[l];
        const bf16* Abuf = (l == 0) ? Xt : Y0;
        for (int d = 0; d < 2; ++d) {
            k_zerof<<<g1((long)B_ * H_), 256, 0, stream>>>(hS, (long)B_ * H_);
            k_zerof<<<g1((long)B_ * H_), 256, 0, stream>>>(cS, (long)B_ * H_);
            k_zerob<<<g1((long)B_ * H_), 256, 0, stream>>>(hSb, (long)B_ * H_);
            for (int c = 0; c < T_ / CHUNK_; ++c) {
                long tstart = (d == 0) ? (long)c * CHUNK_ : (long)(T_ - 1 - c * CHUNK_);
                long ahi = (d == 0) ? (long)B_ * kin : -(long)B_ * kin;
                // batched input projection for CHUNK_ timesteps (off critical path)
                gemm(Abuf + tstart * B_ * kin, kin, ahi, eWih[l][d], kin, G_,
                     nullptr, Gx, G_, (long)B_ * G_, nullptr, nullptr, eBs[l][d],
                     (long)CHUNK_ * B_);
                for (int tl = 0; tl < CHUNK_; ++tl) {
                    long t = (d == 0) ? (long)(c * CHUNK_ + tl) : (long)(T_ - 1 - (c * CHUNK_ + tl));
                    // recurrence: Gstep = Gx[tl] + h*Whh^T
                    gemm(hSb, H_, 0, eWhh[l][d], H_, G_,
                         Gstep, nullptr, G_, 0, nullptr, Gx + (long)tl * B_ * G_, nullptr, B_);
                    bf16* hb2 = (l == 0) ? (Y0 + t * (long)B_ * 1024 + (long)d * 512) : nullptr;
                    gate(Gstep, cS, cS, hS, hSb, hb2, 1024, B_);
                }
            }
            // final hidden into [l0f | l0b | l1f | l1b]
            k_copy_rows_b<<<g1((long)B_ * H_), 256, 0, stream>>>(
                hSb, hidB + (long)(l * 1024 + d * 512), (long)G_, (long)B_ * H_);
        }
    }

    // ================= latent =================
    gemm(hidB, G_, 0, Wmu, G_, LZ_, o_mu, nullptr, LZ_, 0, nullptr, nullptr, h2mub, B_);
    gemm(hidB, G_, 0, Wlv, G_, LZ_, o_lv, nullptr, LZ_, 0, nullptr, nullptr, h2lvb, B_);
    k_reparam<<<g1((long)B_ * LZ_), 256, 0, stream>>>(o_mu, o_lv, eps, zbuf, (long)B_ * LZ_);
    gemm(zbuf, LZ_, 0, Wzi, LZ_, NC_, nullptr, zinB, NC_, 0, nullptr, nullptr, z2inb, B_);
    gemm(zbuf, LZ_, 0, Wzh, LZ_, H_, zhidF, zhidB, H_, 0, nullptr, nullptr, z2hidb, B_);

    // ================= conductor (2-layer LSTM, NBARS steps) =================
    k_copyf<<<g1((long)B_ * H_), 256, 0, stream>>>(ch0, zhidF, (long)B_ * H_);
    k_copyf<<<g1((long)B_ * H_), 256, 0, stream>>>(ch1, zhidF, (long)B_ * H_);
    k_copyb<<<g1((long)B_ * H_), 256, 0, stream>>>(ch0b, zhidB, (long)B_ * H_);
    k_copyb<<<g1((long)B_ * H_), 256, 0, stream>>>(ch1b, zhidB, (long)B_ * H_);
    k_zerof<<<g1((long)B_ * H_), 256, 0, stream>>>(cc0, (long)B_ * H_);
    k_zerof<<<g1((long)B_ * H_), 256, 0, stream>>>(cc1, (long)B_ * H_);
    // constant input projection (z_in is fixed across steps)
    gemm(zinB, NC_, 0, cWih0, NC_, G_, Gc0, nullptr, G_, 0, nullptr, nullptr, cBs0, B_);
    for (int bar = 0; bar < NB_; ++bar) {
        gemm(ch0b, H_, 0, cWhh0, H_, G_, Gstep, nullptr, G_, 0, Gc0, nullptr, nullptr, B_);
        gate(Gstep, cc0, cc0, ch0, ch0b, nullptr, 0, B_);
        gemm(ch0b, H_, 0, cWih1, H_, G_, Gstep, nullptr, G_, 0, nullptr, nullptr, cBs1, B_);
        gemm(ch1b, H_, 0, cWhh1, H_, G_, Gstep, nullptr, G_, 0, Gstep, nullptr, nullptr, B_);
        gate(Gstep, cc1, cc1, ch1, ch1b, embB + (long)bar * B_ * H_, H_, B_);
    }

    // ================= decoder (all 34 bars batched; 8 steps) =================
    {
        long n = (long)NB_ * 2 * B_ * H_;
        k_unpack_dec<<<g1(n), 256, 0, stream>>>(dh0i, dh0, dh1, dh0b, dh1b, n);
        k_unpack_dec<<<g1(n), 256, 0, stream>>>(dc0i, dc0, dc1, (bf16*)nullptr, (bf16*)nullptr, n);
    }
    for (int s = 0; s < 8; ++s) {
        long n = MD_ * (NC_ + H_);
        k_pack_decx<<<g1(n), 256, 0, stream>>>(embB, x, decX, s, n);
        gemm(decX, NC_ + H_, (long)B_ * (NC_ + H_), dWih0, NC_ + H_, G_,
             Gd, nullptr, G_, (long)B_ * G_, nullptr, nullptr, dBs0, MD_);
        gemm(dh0b, H_, (long)B_ * H_, dWhh0, H_, G_,
             Gd, nullptr, G_, (long)B_ * G_, Gd, nullptr, nullptr, MD_);
        gate(Gd, dc0, dc0, dh0, dh0b, nullptr, 0, MD_);
        gemm(dh0b, H_, (long)B_ * H_, dWih1, H_, G_,
             Gd, nullptr, G_, (long)B_ * G_, nullptr, nullptr, dBs1, MD_);
        gemm(dh1b, H_, (long)B_ * H_, dWhh1, H_, G_,
             Gd, nullptr, G_, (long)B_ * G_, Gd, nullptr, nullptr, MD_);
        gate(Gd, dc1, dc1, dh1, dh1b, nullptr, 0, MD_);
        gemm(dh1b, H_, (long)B_ * H_, Wout, H_, NC_,
             Lt, nullptr, NC_, (long)B_ * NC_, nullptr, nullptr, outb, MD_);
        k_softmax<<<g1(MD_ * 32), 256, 0, stream>>>(Lt, o_sm, o_lp, s);
    }

    // ================= passthrough output #5: x =================
    hipMemcpyAsync(o_x, x, sizeof(float) * SZ, hipMemcpyDeviceToDevice, stream);
}
